// Spiking_Oneway_87076166959364
// MI455X (gfx1250) — matrix operand order repaired and checked by reading, not compiled, not run on hardware
//
#include <hip/hip_runtime.h>

typedef __attribute__((ext_vector_type(16))) _Float16 v16h;
typedef __attribute__((ext_vector_type(8)))  _Float16 v8h;
typedef __attribute__((ext_vector_type(8)))  float    v8f;

#define CIN   128
#define COUT  256
#define HH    56
#define WW    56
#define HWSZ  3136      // 56*56
#define TT    4
#define BB    16
#define KTOT  1152      // 9 taps * 128 channels
#define KC    32        // K chunk per WMMA
#define NKC   36        // KTOT / KC
#define BTILE (TT * 16 * KC)   // one LDS buffer: 2048 halves = 4 KB

// workspace layout (bytes)
#define WS_W16    0
#define WS_SCALE  (COUT * KTOT * 2)            // 589824
#define WS_SHIFT  (WS_SCALE + COUT * 4)        // +1024
#define WS_ZPAGE  (WS_SHIFT + COUT * 4)        // +1024 -> 591872 (64B aligned)
#define WS_X16    (WS_ZPAGE + 256)             // 592128 (256B aligned)
#define WS_NEED   ((size_t)WS_X16 + (size_t)BB * TT * HWSZ * CIN * 2)

// ---------------------------------------------------------------------------
// Prep A: weights fp32 [Cout][Cin][3][3] -> fp16 matrix [Cout][tap][Cin];
// fold BN + bias into per-channel scale/shift; zero the guard page.
// ---------------------------------------------------------------------------
__global__ void prep_kernel(const float* __restrict__ conv_w,
                            const float* __restrict__ conv_b,
                            const float* __restrict__ gamma,
                            const float* __restrict__ beta,
                            const float* __restrict__ run_mean,
                            const float* __restrict__ run_var,
                            _Float16* __restrict__ W16,
                            float* __restrict__ scale,
                            float* __restrict__ shift,
                            _Float16* __restrict__ zpage) {
  int gid = blockIdx.x * blockDim.x + threadIdx.x;
  if (gid < COUT * KTOT) {
    const int cout = gid / KTOT;
    const int rem  = gid - cout * KTOT;
    const int tap  = rem / CIN;          // 0..8
    const int ci   = rem - tap * CIN;    // 0..127
    W16[gid] = (_Float16)conv_w[(cout * CIN + ci) * 9 + tap];
  }
  if (gid < COUT) {
    float inv = gamma[gid] / sqrtf(run_var[gid] + 1e-5f);
    scale[gid] = inv;
    shift[gid] = conv_b[gid] * inv + beta[gid] - run_mean[gid] * inv;
  }
  if (gid < 128) zpage[gid] = (_Float16)0.0f;  // 256-B zero guard page
}

// ---------------------------------------------------------------------------
// Prep B: transpose x fp32 NCHW -> fp16 NHWC: x16[bt][h][w][ci].
// Each thread: 8 strided b32 reads (one clause) -> one b128 write.
// ---------------------------------------------------------------------------
__global__ void prep_x16_kernel(const float* __restrict__ x,
                                _Float16* __restrict__ x16) {
  const int gid = blockIdx.x * blockDim.x + threadIdx.x;   // 64*3136*16 total
  const int pos = gid >> 4;            // (bt, hw)
  const int cig = gid & 15;            // channel group of 8
  const int bt  = pos / HWSZ;
  const int hw  = pos - bt * HWSZ;
  const float* src = x + ((size_t)bt * CIN + cig * 8) * HWSZ + hw;
  v8h pack;
#pragma unroll
  for (int e = 0; e < 8; ++e)
    pack[e] = (_Float16)src[(size_t)e * HWSZ];
  *reinterpret_cast<v8h*>(x16 + (size_t)pos * CIN + cig * 8) = pack;
}

// ---------------------------------------------------------------------------
// Fast path: implicit-GEMM conv (WMMA f16->f32) + BN + T=4 spiking recursion.
// WG = 8 waves covers all 256 Cout x 16 spatial x 4 t. Staging = ONE
// global_load_async_to_lds_b128 per thread per K-chunk (ASYNCcnt path),
// boundary handled by redirecting src to a zero page. Double-buffered LDS.
// ---------------------------------------------------------------------------
__global__ void __launch_bounds__(256)
conv_spike_async_kernel(const _Float16* __restrict__ x16,
                        const _Float16* __restrict__ W16,
                        const float* __restrict__ scale,
                        const float* __restrict__ shift,
                        const float* __restrict__ act_alpha,
                        const _Float16* __restrict__ zpage,
                        float* __restrict__ out) {
  __shared__ __align__(64) _Float16 Bt[2 * BTILE];   // 2 x 4 KB

  const int tid  = threadIdx.x;
  const int lane = tid & 31;
  const int wave = tid >> 5;
  const int grp  = lane >> 4;
  const int ln   = lane & 15;

  const int nBase = blockIdx.x * 16;               // over B*H*W = 50176

  // staging decomposition: 2048 halves / 256 threads = 8 halves = 16 B each
  const int tn  = tid >> 2;        // (t, n) pair: 0..63
  const int st  = tn >> 4;         // timestep
  const int sn  = tn & 15;         // column within tile
  const int kk0 = (tid & 3) * 8;   // 8 consecutive channels
  const int gn  = nBase + sn;
  const int sb  = gn / HWSZ;
  const int shw = gn - sb * HWSZ;
  const int sh  = shw / WW;
  const int sw  = shw - sh * WW;
  const _Float16* xrow = x16 + (size_t)(sb * TT + st) * HWSZ * CIN;

  // two A fragments per wave (rows wave*16+ln and +128), contiguous k = 16*grp + s per lane
  const int arow0 = wave * 16 + ln;
  const _Float16* aBase0 = W16 + (size_t)arow0 * KTOT + grp * 16;
  const _Float16* aBase1 = aBase0 + (size_t)128 * KTOT;

  // stage one K-chunk: one async b128 copy straight into LDS per thread
  auto stage_async = [&](int kchunk, _Float16* dst) {
    const int tap = kchunk >> 2;
    const int cc  = kchunk & 3;
    const int kh  = tap / 3;
    const int kw  = tap - kh * 3;
    const int ih  = sh + kh - 1;
    const int iw  = sw + kw - 1;
    const bool valid = ((unsigned)ih < (unsigned)HH) &
                       ((unsigned)iw < (unsigned)WW);
    const _Float16* src = valid
        ? xrow + ((ptrdiff_t)(ih * WW + iw) * CIN + cc * 32 + kk0)
        : zpage;
    // low 32 bits of a flat LDS address == LDS byte offset (ISA 10.2)
    const unsigned ldsoff =
        (unsigned)(uintptr_t)(dst + st * (16 * KC) + sn * KC + kk0);
    asm volatile("global_load_async_to_lds_b128 %0, %1, off"
                 :: "v"(ldsoff), "v"(src) : "memory");
  };

  v8f acc[2][TT] = {};

  stage_async(0, Bt);                      // prime buffer 0
  for (int kchunk = 0; kchunk < NKC; ++kchunk) {
    asm volatile("s_wait_asynccnt 0x0" ::: "memory");
    __syncthreads();
    _Float16* cur = Bt + (kchunk & 1) * BTILE;
    if (kchunk + 1 < NKC)                  // async copy overlaps WMMAs below
      stage_async(kchunk + 1, Bt + ((kchunk + 1) & 1) * BTILE);

    const v8h* ap0 = reinterpret_cast<const v8h*>(aBase0 + (size_t)kchunk * KC);
    const v8h* ap1 = reinterpret_cast<const v8h*>(aBase1 + (size_t)kchunk * KC);
    const v8h a00 = ap0[0], a01 = ap0[1];
    const v8h a10 = ap1[0], a11 = ap1[1];
    if (kchunk + 2 < NKC) {
      __builtin_prefetch(aBase0 + (size_t)(kchunk + 2) * KC, 0, 1);
      __builtin_prefetch(aBase1 + (size_t)(kchunk + 2) * KC, 0, 1);
    }
    v16h a0, a1;
#pragma unroll
    for (int i = 0; i < 8; ++i) {
      a0[i] = a00[i]; a0[i + 8] = a01[i];
      a1[i] = a10[i]; a1[i + 8] = a11[i];
    }

#pragma unroll
    for (int t = 0; t < TT; ++t) {
      const v16h b = *reinterpret_cast<const v16h*>(
          cur + t * (16 * KC) + ln * KC + grp * 16);
      acc[0][t] = __builtin_amdgcn_wmma_f32_16x16x32_f16(
          false, a0, false, b, (short)0, acc[0][t], false, false);
      acc[1][t] = __builtin_amdgcn_wmma_f32_16x16x32_f16(
          false, a1, false, b, (short)0, acc[1][t], false, false);
    }
  }

  // epilogue: BN fold + spiking recursion in registers
  const float thr = act_alpha[0];
  const int onn = nBase + ln;
  const int ob  = onn / HWSZ;
  const int ohw = onn - ob * HWSZ;

#pragma unroll
  for (int half = 0; half < 2; ++half) {
#pragma unroll
    for (int i = 0; i < 8; ++i) {
      const int cout = half * 128 + wave * 16 + grp * 8 + i;
      const float sc = scale[cout];
      const float sf = shift[cout];
      float mem = 0.5f * thr;
      float ss  = 0.0f;
      float* obase = out + ((size_t)ob * TT * COUT + cout) * HWSZ + ohw;
#pragma unroll
      for (int t = 0; t < TT; ++t) {
        const float y = acc[half][t][i] * sc + sf;
        mem += y;
        const float spike = (mem >= thr) ? 1.0f : 0.0f;
        mem -= thr * spike;
        ss  += spike;
        const float inh = ((mem <= -0.001f) && (ss > 0.0f)) ? 1.0f : 0.0f;
        mem += thr * inh;
        ss  -= inh;
        obase[(size_t)t * COUT * HWSZ] = (spike - inh) * thr;
      }
    }
  }
}

// ---------------------------------------------------------------------------
// Fallback (small workspace): round-3 gather kernel, f32 x in place.
// ---------------------------------------------------------------------------
__global__ void __launch_bounds__(256)
conv_spike_fallback_kernel(const float* __restrict__ x,
                           const _Float16* __restrict__ W16,
                           const float* __restrict__ scale,
                           const float* __restrict__ shift,
                           const float* __restrict__ act_alpha,
                           float* __restrict__ out) {
  __shared__ __align__(64) _Float16 Bt[2 * BTILE];

  const int tid  = threadIdx.x;
  const int lane = tid & 31;
  const int wave = tid >> 5;
  const int grp  = lane >> 4;
  const int ln   = lane & 15;
  const int nBase = blockIdx.x * 16;

  const int tn  = tid >> 2;
  const int st  = tn >> 4;
  const int sn  = tn & 15;
  const int kk0 = (tid & 3) * 8;
  const int gn  = nBase + sn;
  const int sb  = gn / HWSZ;
  const int shw = gn - sb * HWSZ;
  const int sh  = shw / WW;
  const int sw  = shw - sh * WW;
  const size_t xbase0 = (size_t)(sb * TT + st) * CIN * HWSZ;

  const int arow0 = wave * 16 + ln;
  const _Float16* aBase0 = W16 + (size_t)arow0 * KTOT + grp * 16;
  const _Float16* aBase1 = aBase0 + (size_t)128 * KTOT;

  auto stage_chunk = [&](int kchunk, _Float16* dst) {
    const int tap = kchunk >> 2;
    const int cc  = kchunk & 3;
    const int kh  = tap / 3;
    const int kw  = tap - kh * 3;
    const int ih  = sh + kh - 1;
    const int iw  = sw + kw - 1;
    const bool valid = ((unsigned)ih < (unsigned)HH) &
                       ((unsigned)iw < (unsigned)WW);
    const int ihc = ih < 0 ? 0 : (ih > HH - 1 ? HH - 1 : ih);
    const int iwc = iw < 0 ? 0 : (iw > WW - 1 ? WW - 1 : iw);
    const float msk = valid ? 1.0f : 0.0f;
    const float* xb = x + xbase0 + (ihc * WW + iwc);
    const int ci0 = cc * 32 + kk0;
    float vals[8];
#pragma unroll
    for (int e = 0; e < 8; ++e)
      vals[e] = xb[(size_t)(ci0 + e) * HWSZ];
    v8h pack;
#pragma unroll
    for (int e = 0; e < 8; ++e)
      pack[e] = (_Float16)(vals[e] * msk);
    *reinterpret_cast<v8h*>(dst + st * (16 * KC) + sn * KC + kk0) = pack;
  };

  v8f acc[2][TT] = {};

  stage_chunk(0, Bt);
  for (int kchunk = 0; kchunk < NKC; ++kchunk) {
    __syncthreads();
    _Float16* cur = Bt + (kchunk & 1) * BTILE;
    if (kchunk + 1 < NKC)
      stage_chunk(kchunk + 1, Bt + ((kchunk + 1) & 1) * BTILE);

    const v8h* ap0 = reinterpret_cast<const v8h*>(aBase0 + (size_t)kchunk * KC);
    const v8h* ap1 = reinterpret_cast<const v8h*>(aBase1 + (size_t)kchunk * KC);
    const v8h a00 = ap0[0], a01 = ap0[1];
    const v8h a10 = ap1[0], a11 = ap1[1];
    v16h a0, a1;
#pragma unroll
    for (int i = 0; i < 8; ++i) {
      a0[i] = a00[i]; a0[i + 8] = a01[i];
      a1[i] = a10[i]; a1[i + 8] = a11[i];
    }
#pragma unroll
    for (int t = 0; t < TT; ++t) {
      const v16h b = *reinterpret_cast<const v16h*>(
          cur + t * (16 * KC) + ln * KC + grp * 16);
      acc[0][t] = __builtin_amdgcn_wmma_f32_16x16x32_f16(
          false, a0, false, b, (short)0, acc[0][t], false, false);
      acc[1][t] = __builtin_amdgcn_wmma_f32_16x16x32_f16(
          false, a1, false, b, (short)0, acc[1][t], false, false);
    }
  }

  const float thr = act_alpha[0];
  const int onn = nBase + ln;
  const int ob  = onn / HWSZ;
  const int ohw = onn - ob * HWSZ;

#pragma unroll
  for (int half = 0; half < 2; ++half) {
#pragma unroll
    for (int i = 0; i < 8; ++i) {
      const int cout = half * 128 + wave * 16 + grp * 8 + i;
      const float sc = scale[cout];
      const float sf = shift[cout];
      float mem = 0.5f * thr;
      float ss  = 0.0f;
      float* obase = out + ((size_t)ob * TT * COUT + cout) * HWSZ + ohw;
#pragma unroll
      for (int t = 0; t < TT; ++t) {
        const float y = acc[half][t][i] * sc + sf;
        mem += y;
        const float spike = (mem >= thr) ? 1.0f : 0.0f;
        mem -= thr * spike;
        ss  += spike;
        const float inh = ((mem <= -0.001f) && (ss > 0.0f)) ? 1.0f : 0.0f;
        mem += thr * inh;
        ss  -= inh;
        obase[(size_t)t * COUT * HWSZ] = (spike - inh) * thr;
      }
    }
  }
}

// ---------------------------------------------------------------------------
extern "C" void kernel_launch(void* const* d_in, const int* in_sizes, int n_in,
                              void* d_out, int out_size, void* d_ws, size_t ws_size,
                              hipStream_t stream) {
  const float* x         = (const float*)d_in[0];
  const float* conv_w    = (const float*)d_in[1];
  const float* conv_b    = (const float*)d_in[2];
  const float* gamma     = (const float*)d_in[3];
  const float* beta      = (const float*)d_in[4];
  const float* run_mean  = (const float*)d_in[5];
  const float* run_var   = (const float*)d_in[6];
  const float* act_alpha = (const float*)d_in[7];
  float* out = (float*)d_out;

  char* ws = (char*)d_ws;
  _Float16* W16   = (_Float16*)(ws + WS_W16);
  float*    scale = (float*)(ws + WS_SCALE);
  float*    shift = (float*)(ws + WS_SHIFT);
  _Float16* zpage = (_Float16*)(ws + WS_ZPAGE);
  _Float16* x16   = (_Float16*)(ws + WS_X16);

  const int prepN = COUT * KTOT;
  prep_kernel<<<(prepN + 255) / 256, 256, 0, stream>>>(
      conv_w, conv_b, gamma, beta, run_mean, run_var, W16, scale, shift, zpage);

  dim3 grid((BB * HWSZ) / 16, 1, 1);   // 3136 spatial tiles, all Cout per WG

  if (ws_size >= WS_NEED) {
    // fast path: NHWC f16 transpose once, then async-to-LDS staging
    const int xN = BB * TT * HWSZ * (CIN / 8);   // threads (8 ch each)
    prep_x16_kernel<<<(xN + 255) / 256, 256, 0, stream>>>(x, x16);
    conv_spike_async_kernel<<<grid, 256, 0, stream>>>(
        x16, W16, scale, shift, act_alpha, zpage, out);
  } else {
    conv_spike_fallback_kernel<<<grid, 256, 0, stream>>>(
        x, W16, scale, shift, act_alpha, out);
  }
}
